// DynamicConv_14173392077335
// MI455X (gfx1250) — matrix operand order repaired and checked by reading, not compiled, not run on hardware
//
#include <hip/hip_runtime.h>
#include <hip/hip_bf16.h>

typedef __bf16 bf16;
typedef __attribute__((ext_vector_type(16))) __bf16 v16bf;
typedef __attribute__((ext_vector_type(8)))  float  v8f;

constexpr int kH    = 256;    // hidden
constexpr int kD    = 64;     // dynamic dim
constexpr int kT    = 49;     // tokens
constexpr int kB    = 4800;   // boxes
constexpr int kNP   = kH * kD;        // 16384
constexpr int k2NP  = 2 * kNP;        // 32768
constexpr int kNOUT = kH * kT;        // 12544
constexpr float kEPS = 1e-5f;

// ---------------------------------------------------------------------------
// WMMA fragment helpers (CDNA5 16x16x32 bf16, wave32; A/B in this code's contiguous k order, C/D per ISA 7.12.2)
// A (16x32, MxK): lane L -> row = L%16, half = L/16;
//   element e -> K = half*16 + e
// B (32x16, KxN): lane L -> col = L%16, half = L/16; element e -> K = half*16+e
// C/D (16x16 f32): lane L -> col = L%16, half = L/16; vgpr r -> row = 8*half+r
// ---------------------------------------------------------------------------
__device__ __forceinline__ int a_koff(int e, int hlf) {
  return ((e >> 3) << 4) + hlf * 8 + (e & 7);
}

__device__ __forceinline__ v16bf load_a_frag(const bf16* base, int stride,
                                             int row, int k0, int lane) {
  const int hlf = lane >> 4;
  const bf16* p = base + (size_t)row * stride + k0;
  v16bf a;
#pragma unroll
  for (int e = 0; e < 16; ++e) a[e] = p[hlf * 16 + e];
  return a;
}

__device__ __forceinline__ v16bf load_b_frag(const bf16* base, int stride,
                                             int ncol, int k0, int lane) {
  const int hlf = lane >> 4;
  const bf16* p = base + (size_t)ncol * stride + k0 + hlf * 16;
  v16bf b;
#pragma unroll
  for (int e = 0; e < 16; ++e) b[e] = p[e];
  return b;
}

__device__ __forceinline__ v8f wmma_bf16(v16bf a, v16bf b, v8f c) {
  return __builtin_amdgcn_wmma_f32_16x16x32_bf16(false, a, false, b,
                                                 (short)0, c, false, false);
}

// ---------------------------------------------------------------------------
// CDNA5 async global->LDS DMA (ASYNCcnt-tracked). LDS offset = low 32 bits of
// the flat pointer (LDS aperture, ISA 10.2).
// ---------------------------------------------------------------------------
__device__ __forceinline__ void async_load_b128(uint32_t lds_off,
                                                const void* gptr) {
  asm volatile("global_load_async_to_lds_b128 %0, %1, off"
               :: "v"(lds_off), "v"(gptr)
               : "memory");
}
__device__ __forceinline__ void wait_async0() {
  asm volatile("s_wait_asynccnt 0x0" ::: "memory");
}
__device__ __forceinline__ uint32_t lds_off32(const void* p) {
  return (uint32_t)(uintptr_t)p;
}

// ---------------------------------------------------------------------------
// Kernel 0: fp32 -> bf16 conversion (grid-stride)
// ---------------------------------------------------------------------------
__global__ void cvt_f32_bf16(const float* __restrict__ src,
                             bf16* __restrict__ dst, long n) {
  long i = (long)blockIdx.x * blockDim.x + threadIdx.x;
  long stride = (long)gridDim.x * blockDim.x;
  for (; i < n; i += stride) dst[i] = (bf16)src[i];
}

// ---------------------------------------------------------------------------
// Kernel 1: params = pro(4800x256) @ dyn_w^T(256x32768) + dyn_b, bf16 out,
// stored PRE-TRANSPOSED per box so the consumer can DMA contiguously:
//   p1 (p = h*64+d, h is K)  -> paramsT[b][ d*256 + h ]        ([d][h])
//   p2 (p = 16384 + d*256+h) -> paramsT[b][ 16384 + h*64 + d ] ([h][d])
// The B-operand fragment reads arbitrary dyn_w rows per lane, so we permute
// the 128 block columns so that stores stay 32B-coalesced along the fast
// output index (h for p1, d for p2).
// Block = 64x128 tile, 256 threads = 8 waves, each wave a 32x32 sub-tile.
// ---------------------------------------------------------------------------
__global__ __launch_bounds__(256) void k_params_gemm(
    const bf16* __restrict__ proB,     // [4800][256]
    const bf16* __restrict__ dynW,     // [32768][256]
    const float* __restrict__ dynBias, // [32768]
    bf16* __restrict__ paramsT)        // [4800][32768] (transposed layout)
{
  const int lane = threadIdx.x & 31, w = threadIdx.x >> 5;
  const int hlf = lane >> 4, l16 = lane & 15;
  const int m0 = blockIdx.x * 64 + (w >> 2) * 32;
  const int cbase = (w & 3) * 32;               // wave's column pair base
  const bool is_p2 = (blockIdx.y >= 128);
  const int yy = is_p2 ? (int)blockIdx.y - 128 : (int)blockIdx.y;

  int nrow[2]; int ooff[2];
#pragma unroll
  for (int ni = 0; ni < 2; ++ni) {
    const int c = cbase + ni * 16 + l16;        // 0..127 within block
    const int hi = c >> 6, lo = c & 63;
    if (!is_p2) {
      const int d = (yy & 31) * 2 + hi;         // block: 2 d x 64 h
      const int h = (yy >> 5) * 64 + lo;
      nrow[ni] = h * kD + d;                    // dyn_w row (original n)
      ooff[ni] = d * kH + h;                    // transposed store offset
    } else {
      const int h = yy * 2 + hi;                // block: 2 h x 64 d
      const int d = lo;
      nrow[ni] = kNP + d * kH + h;
      ooff[ni] = kNP + h * kD + d;
    }
  }

  v8f acc[2][2] = {};
  for (int k0 = 0; k0 < kH; k0 += 32) {
    v16bf a0 = load_a_frag(proB, kH, m0 + l16,      k0, lane);
    v16bf a1 = load_a_frag(proB, kH, m0 + 16 + l16, k0, lane);
    v16bf b0 = load_b_frag(dynW, kH, nrow[0], k0, lane);
    v16bf b1 = load_b_frag(dynW, kH, nrow[1], k0, lane);
    acc[0][0] = wmma_bf16(a0, b0, acc[0][0]);
    acc[0][1] = wmma_bf16(a0, b1, acc[0][1]);
    acc[1][0] = wmma_bf16(a1, b0, acc[1][0]);
    acc[1][1] = wmma_bf16(a1, b1, acc[1][1]);
  }
#pragma unroll
  for (int mi = 0; mi < 2; ++mi)
#pragma unroll
    for (int ni = 0; ni < 2; ++ni) {
      const float bias = dynBias[nrow[ni]];
#pragma unroll
      for (int r = 0; r < 8; ++r) {
        const int m = m0 + mi * 16 + hlf * 8 + r;
        paramsT[(size_t)m * k2NP + ooff[ni]] = (bf16)(acc[mi][ni][r] + bias);
      }
    }
}

// ---------------------------------------------------------------------------
// Kernel 2: fully fused per-box middle section (one workgroup per box).
//   x1 = feats(49x256) @ p1(256x64); LN(D)+relu
//   x  = x1(49x64)     @ p2(64x256); LN(H)+relu
//   masks = sigmoid(2-layer projections of pro[b])
//   fc/fr = relu(LN(x * mask));  stored bf16 to workspace
// p1/p2 arrive pre-transposed -> staged into LDS with async-to-LDS DMA.
// ---------------------------------------------------------------------------
__global__ __launch_bounds__(256) void k_fused_box(
    const float* __restrict__ roi,   // [49][4800][256]
    const float* __restrict__ pro,   // [4800][256]
    const bf16* __restrict__ paramsT,// [4800][32768] transposed
    const float* __restrict__ n1_g, const float* __restrict__ n1_b,
    const float* __restrict__ n2_g, const float* __restrict__ n2_b,
    const float* __restrict__ pc1_w, const float* __restrict__ pc1_b,
    const float* __restrict__ pc2_w, const float* __restrict__ pc2_b,
    const float* __restrict__ pr1_w, const float* __restrict__ pr1_b,
    const float* __restrict__ pr2_w, const float* __restrict__ pr2_b,
    const float* __restrict__ n3c_g, const float* __restrict__ n3c_b,
    const float* __restrict__ n3r_g, const float* __restrict__ n3r_b,
    bf16* __restrict__ fcB,          // [4800][12544]
    bf16* __restrict__ frB)          // [4800][12544]
{
  __shared__ __align__(16) char smem[57344];
  __shared__ float pro_s[256];
  __shared__ float h1c[32], h1r[32];
  __shared__ float mask_c[256], mask_r[256];

  bf16*  featsA = (bf16*)smem;             // [64][256] bf16 (phase A)
  bf16*  p1T    = (bf16*)(smem + 32768);   // [64][128] bf16 chunk (phase A)
  float* x1f    = (float*)smem;            // [64][64]  f32  (after GEMM1)
  bf16*  x1A    = (bf16*)(smem + 16384);   // [64][64]  bf16
  bf16*  p2T    = (bf16*)(smem + 24576);   // [256][64] bf16
  float* xf     = (float*)smem;            // [49][256] f32  (after GEMM2)

  const int tid = threadIdx.x;
  const int lane = tid & 31, w = tid >> 5;
  const int hlf = lane >> 4, l16 = lane & 15;
  const int b = blockIdx.x;
  const size_t pbase = (size_t)b * k2NP;

  // ---- load pro and feats (zero-pad rows 49..63) ----
  pro_s[tid] = pro[(size_t)b * kH + tid];
  for (int idx = tid; idx < 64 * 256; idx += 256) {
    const int t = idx >> 8, h = idx & 255;
    featsA[idx] = (t < kT) ? (bf16)roi[((size_t)t * kB + b) * kH + h] : (bf16)0.f;
  }

  // ---- GEMM1: 16 tiles (4m x 4n), wave -> 1 mtile x 2 ntiles ----
  const int mt  = w & 3;
  const int np0 = (w >> 2) * 2;
  v8f acc1[2] = {};
  for (int chunk = 0; chunk < 2; ++chunk) {
    const int hbase = chunk * 128;
    __syncthreads();
    // stage p1T[d][hh] via async DMA: source rows are contiguous (p1 stored
    // [d][h]); 1024 x 16B chunks, 4 per thread.
    for (int u = tid; u < 1024; u += 256) {
      const int d = u >> 4, j = u & 15;
      async_load_b128(lds_off32(p1T + d * 128 + j * 8),
                      paramsT + pbase + (size_t)d * kH + hbase + j * 8);
    }
    wait_async0();
    __syncthreads();
    for (int k0 = 0; k0 < 128; k0 += 32) {
      v16bf a  = load_a_frag(featsA, 256, mt * 16 + l16, hbase + k0, lane);
      v16bf b0 = load_b_frag(p1T, 128, (np0    ) * 16 + l16, k0, lane);
      v16bf b1 = load_b_frag(p1T, 128, (np0 + 1) * 16 + l16, k0, lane);
      acc1[0] = wmma_bf16(a, b0, acc1[0]);
      acc1[1] = wmma_bf16(a, b1, acc1[1]);
    }
  }
  __syncthreads();                       // done reading featsA/p1T
#pragma unroll
  for (int ni = 0; ni < 2; ++ni)
#pragma unroll
    for (int r = 0; r < 8; ++r)
      x1f[(mt * 16 + hlf * 8 + r) * 64 + (np0 + ni) * 16 + l16] = acc1[ni][r];
  __syncthreads();

  // ---- stage p2T (fully contiguous in transposed params) via async DMA ----
  for (int u = tid; u < 4096; u += 256)
    async_load_b128(lds_off32(p2T + u * 8),
                    paramsT + pbase + kNP + (size_t)u * 8);

  // ---- LN1 over D=64 + relu -> x1A (bf16), wave-per-row ----
  for (int t = w; t < kT; t += 8) {
    const float v0 = x1f[t * 64 + lane];
    const float v1 = x1f[t * 64 + lane + 32];
    float s = v0 + v1;
#pragma unroll
    for (int off = 16; off >= 1; off >>= 1) s += __shfl_xor(s, off, 32);
    const float mu = s * (1.f / 64.f);
    const float d0 = v0 - mu, d1 = v1 - mu;
    float q = d0 * d0 + d1 * d1;
#pragma unroll
    for (int off = 16; off >= 1; off >>= 1) q += __shfl_xor(q, off, 32);
    const float rs = rsqrtf(q * (1.f / 64.f) + kEPS);
    x1A[t * 64 + lane]      = (bf16)fmaxf(d0 * rs * n1_g[lane]      + n1_b[lane],      0.f);
    x1A[t * 64 + lane + 32] = (bf16)fmaxf(d1 * rs * n1_g[lane + 32] + n1_b[lane + 32], 0.f);
  }
  for (int idx = tid; idx < (64 - kT) * 64; idx += 256)
    x1A[kT * 64 + idx] = (bf16)0.f;
  // mask stage 1 (two 32x256 matvecs)
  if (tid < 32) {
    float a = pc1_b[tid];
    for (int h = 0; h < 256; ++h) a += pro_s[h] * pc1_w[tid * 256 + h];
    h1c[tid] = a;
  } else if (tid < 64) {
    const int j = tid - 32;
    float a = pr1_b[j];
    for (int h = 0; h < 256; ++h) a += pro_s[h] * pr1_w[j * 256 + h];
    h1r[j] = a;
  }
  wait_async0();                         // p2T resident
  __syncthreads();

  // mask stage 2 (sigmoid)
  {
    float ac = pc2_b[tid], ar = pr2_b[tid];
    for (int j = 0; j < 32; ++j) {
      ac += h1c[j] * pc2_w[tid * 32 + j];
      ar += h1r[j] * pr2_w[tid * 32 + j];
    }
    mask_c[tid] = 1.f / (1.f + __expf(-ac));
    mask_r[tid] = 1.f / (1.f + __expf(-ar));
  }

  // ---- GEMM2: 64 tiles (4m x 16n), wave -> 4 mtiles x 2 ntiles ----
  const int n2t0 = w * 2;
  v8f acc2[4][2] = {};
  for (int k0 = 0; k0 < 64; k0 += 32) {
    v16bf bb0 = load_b_frag(p2T, 64, (n2t0    ) * 16 + l16, k0, lane);
    v16bf bb1 = load_b_frag(p2T, 64, (n2t0 + 1) * 16 + l16, k0, lane);
#pragma unroll
    for (int mi = 0; mi < 4; ++mi) {
      v16bf a = load_a_frag(x1A, 64, mi * 16 + l16, k0, lane);
      acc2[mi][0] = wmma_bf16(a, bb0, acc2[mi][0]);
      acc2[mi][1] = wmma_bf16(a, bb1, acc2[mi][1]);
    }
  }
  __syncthreads();                       // done reading x1A/p2T -> xf may alias
#pragma unroll
  for (int mi = 0; mi < 4; ++mi)
#pragma unroll
    for (int ni = 0; ni < 2; ++ni)
#pragma unroll
      for (int r = 0; r < 8; ++r) {
        const int t = mi * 16 + hlf * 8 + r;
        if (t < kT) xf[t * 256 + (n2t0 + ni) * 16 + l16] = acc2[mi][ni][r];
      }
  __syncthreads();

  // ---- LN2 over H=256 + relu (in place), wave-per-row ----
  for (int t = w; t < kT; t += 8) {
    float v[8]; float s = 0.f;
#pragma unroll
    for (int j = 0; j < 8; ++j) { v[j] = xf[t * 256 + lane + 32 * j]; s += v[j]; }
#pragma unroll
    for (int off = 16; off >= 1; off >>= 1) s += __shfl_xor(s, off, 32);
    const float mu = s * (1.f / 256.f);
    float q = 0.f;
#pragma unroll
    for (int j = 0; j < 8; ++j) { v[j] -= mu; q += v[j] * v[j]; }
#pragma unroll
    for (int off = 16; off >= 1; off >>= 1) q += __shfl_xor(q, off, 32);
    const float rs = rsqrtf(q * (1.f / 256.f) + kEPS);
#pragma unroll
    for (int j = 0; j < 8; ++j) {
      const int h = lane + 32 * j;
      xf[t * 256 + h] = fmaxf(v[j] * rs * n2_g[h] + n2_b[h], 0.f);
    }
  }
  __syncthreads();

  // ---- fc / fr : mask, LN3, relu, bf16 store ----
  const size_t obase = (size_t)b * kNOUT;
  for (int t = w; t < kT; t += 8) {
    float xv[8];
#pragma unroll
    for (int j = 0; j < 8; ++j) xv[j] = xf[t * 256 + lane + 32 * j];
    // cls branch
    {
      float u[8]; float s = 0.f;
#pragma unroll
      for (int j = 0; j < 8; ++j) { u[j] = xv[j] * mask_c[lane + 32 * j]; s += u[j]; }
#pragma unroll
      for (int off = 16; off >= 1; off >>= 1) s += __shfl_xor(s, off, 32);
      const float mu = s * (1.f / 256.f);
      float q = 0.f;
#pragma unroll
      for (int j = 0; j < 8; ++j) { u[j] -= mu; q += u[j] * u[j]; }
#pragma unroll
      for (int off = 16; off >= 1; off >>= 1) q += __shfl_xor(q, off, 32);
      const float rs = rsqrtf(q * (1.f / 256.f) + kEPS);
#pragma unroll
      for (int j = 0; j < 8; ++j) {
        const int h = lane + 32 * j;
        fcB[obase + t * 256 + h] =
            (bf16)fmaxf(u[j] * rs * n3c_g[h] + n3c_b[h], 0.f);
      }
    }
    // reg branch
    {
      float u[8]; float s = 0.f;
#pragma unroll
      for (int j = 0; j < 8; ++j) { u[j] = xv[j] * mask_r[lane + 32 * j]; s += u[j]; }
#pragma unroll
      for (int off = 16; off >= 1; off >>= 1) s += __shfl_xor(s, off, 32);
      const float mu = s * (1.f / 256.f);
      float q = 0.f;
#pragma unroll
      for (int j = 0; j < 8; ++j) { u[j] -= mu; q += u[j] * u[j]; }
#pragma unroll
      for (int off = 16; off >= 1; off >>= 1) q += __shfl_xor(q, off, 32);
      const float rs = rsqrtf(q * (1.f / 256.f) + kEPS);
#pragma unroll
      for (int j = 0; j < 8; ++j) {
        const int h = lane + 32 * j;
        frB[obase + t * 256 + h] =
            (bf16)fmaxf(u[j] * rs * n3r_g[h] + n3r_b[h], 0.f);
      }
    }
  }
}

// ---------------------------------------------------------------------------
// Kernel 3: out = relu(LN(X(4800x12544) @ W^T(12544x256) + bias))
// 128 threads = 4 waves; each wave owns 16 rows x all 256 cols (16 accums),
// so LN reduces within a 16-lane half via shfl_xor. No LDS needed.
// ---------------------------------------------------------------------------
__global__ __launch_bounds__(128) void k_out_gemm(
    const bf16* __restrict__ X,     // [4800][12544]
    const bf16* __restrict__ Wt,    // [256][12544]
    const float* __restrict__ bias, // [256]
    const float* __restrict__ g,    // [256]
    const float* __restrict__ bv,   // [256]
    float* __restrict__ out)        // [4800][256]
{
  const int lane = threadIdx.x & 31, w = threadIdx.x >> 5;
  const int hlf = lane >> 4, l16 = lane & 15;
  const int m0 = blockIdx.x * 64 + w * 16;

  v8f acc[16] = {};
  for (int k0 = 0; k0 < kNOUT; k0 += 32) {
    v16bf a = load_a_frag(X, kNOUT, m0 + l16, k0, lane);
#pragma unroll
    for (int nt = 0; nt < 16; ++nt) {
      v16bf bb = load_b_frag(Wt, kNOUT, nt * 16 + l16, k0, lane);
      acc[nt] = wmma_bf16(a, bb, acc[nt]);
    }
  }
#pragma unroll
  for (int r = 0; r < 8; ++r) {
    const int m = m0 + hlf * 8 + r;
    float vals[16]; float s = 0.f;
#pragma unroll
    for (int nt = 0; nt < 16; ++nt) {
      vals[nt] = acc[nt][r] + bias[nt * 16 + l16];
      s += vals[nt];
    }
#pragma unroll
    for (int off = 8; off >= 1; off >>= 1) s += __shfl_xor(s, off, 32);
    const float mu = s * (1.f / 256.f);
    float q = 0.f;
#pragma unroll
    for (int nt = 0; nt < 16; ++nt) {
      const float d = vals[nt] - mu; q += d * d;
    }
#pragma unroll
    for (int off = 8; off >= 1; off >>= 1) q += __shfl_xor(q, off, 32);
    const float rs = rsqrtf(q * (1.f / 256.f) + kEPS);
#pragma unroll
    for (int nt = 0; nt < 16; ++nt) {
      const int n = nt * 16 + l16;
      out[(size_t)m * kH + n] =
          fmaxf((vals[nt] - mu) * rs * g[n] + bv[n], 0.f);
    }
  }
}

// ---------------------------------------------------------------------------
extern "C" void kernel_launch(void* const* d_in, const int* in_sizes, int n_in,
                              void* d_out, int out_size, void* d_ws, size_t ws_size,
                              hipStream_t stream) {
  const float* pro   = (const float*)d_in[0];
  const float* roi   = (const float*)d_in[1];
  const float* dyn_w = (const float*)d_in[2];
  const float* dyn_b = (const float*)d_in[3];
  const float* n1_g  = (const float*)d_in[4];
  const float* n1_b  = (const float*)d_in[5];
  const float* n2_g  = (const float*)d_in[6];
  const float* n2_b  = (const float*)d_in[7];
  const float* pc1_w = (const float*)d_in[8];
  const float* pc1_b = (const float*)d_in[9];
  const float* pc2_w = (const float*)d_in[10];
  const float* pc2_b = (const float*)d_in[11];
  const float* n3c_g = (const float*)d_in[12];
  const float* n3c_b = (const float*)d_in[13];
  const float* pr1_w = (const float*)d_in[14];
  const float* pr1_b = (const float*)d_in[15];
  const float* pr2_w = (const float*)d_in[16];
  const float* pr2_b = (const float*)d_in[17];
  const float* n3r_g = (const float*)d_in[18];
  const float* n3r_b = (const float*)d_in[19];
  const float* oc_w  = (const float*)d_in[20];
  const float* oc_b  = (const float*)d_in[21];
  const float* n4c_g = (const float*)d_in[22];
  const float* n4c_b = (const float*)d_in[23];
  const float* or_w  = (const float*)d_in[24];
  const float* or_b  = (const float*)d_in[25];
  const float* n4r_g = (const float*)d_in[26];
  const float* n4r_b = (const float*)d_in[27];
  (void)in_sizes; (void)n_in; (void)out_size; (void)ws_size;

  // Workspace layout (all sizes multiples of 256 B)
  char* ws = (char*)d_ws;
  bf16* dynW_bf = (bf16*)ws;  ws += (size_t)k2NP * kH * 2;      // 16,777,216
  bf16* ocW_bf  = (bf16*)ws;  ws += (size_t)kH * kNOUT * 2;     //  6,422,528
  bf16* orW_bf  = (bf16*)ws;  ws += (size_t)kH * kNOUT * 2;     //  6,422,528
  bf16* proB_bf = (bf16*)ws;  ws += (size_t)kB * kH * 2;        //  2,457,600
  bf16* paramsT = (bf16*)ws;  ws += (size_t)kB * k2NP * 2;      // 314,572,800
  bf16* fcB     = (bf16*)ws;  ws += (size_t)kB * kNOUT * 2;     // 120,422,400
  bf16* frB     = (bf16*)ws;  ws += (size_t)kB * kNOUT * 2;     // 120,422,400

  // Stage 0: weight / activation bf16 conversion
  cvt_f32_bf16<<<2048, 256, 0, stream>>>(dyn_w, dynW_bf, (long)k2NP * kH);
  cvt_f32_bf16<<<2048, 256, 0, stream>>>(oc_w,  ocW_bf,  (long)kH * kNOUT);
  cvt_f32_bf16<<<2048, 256, 0, stream>>>(or_w,  orW_bf,  (long)kH * kNOUT);
  cvt_f32_bf16<<<1024, 256, 0, stream>>>(pro,   proB_bf, (long)kB * kH);

  // Stage 1: dynamic-parameter GEMM (4800 x 32768 x 256), transposed store
  k_params_gemm<<<dim3(kB / 64, k2NP / 128), 256, 0, stream>>>(
      proB_bf, dynW_bf, dyn_b, paramsT);

  // Stage 2: fused per-box dynamic conv + norms + masks
  k_fused_box<<<kB, 256, 0, stream>>>(
      roi, pro, paramsT, n1_g, n1_b, n2_g, n2_b,
      pc1_w, pc1_b, pc2_w, pc2_b, pr1_w, pr1_b, pr2_w, pr2_b,
      n3c_g, n3c_b, n3r_g, n3r_b, fcB, frB);

  // Stage 3: output GEMMs (4800 x 256 x 12544) + LN + relu
  float* out = (float*)d_out;
  k_out_gemm<<<kB / 64, 128, 0, stream>>>(fcB, ocW_bf, oc_b, n4c_g, n4c_b, out);
  k_out_gemm<<<kB / 64, 128, 0, stream>>>(frB, orW_bf, or_b, n4r_g, n4r_b,
                                          out + (size_t)kB * kH);
}
